// Clustering_74062416053197
// MI455X (gfx1250) — matrix operand order repaired and checked by reading, not compiled, not run on hardware
//
#include <hip/hip_runtime.h>
#include <hip/hip_bf16.h>

// ---------------------------------------------------------------------------
// Types for CDNA5 WMMA (wave32): v_wmma_f32_16x16x32_f16
// ---------------------------------------------------------------------------
typedef _Float16 half4_t __attribute__((ext_vector_type(4)));
typedef _Float16 half8_t __attribute__((ext_vector_type(8)));
typedef _Float16 v16h    __attribute__((ext_vector_type(16)));
typedef float    v8f     __attribute__((ext_vector_type(8)));

union V16HU { v16h v; half8_t h8[2]; };

#define TILE_M 128
#define TILE_N 128
#define TILE_K 32
#define LDH    40   // padded LDS row stride in halves (80B: 16B aligned, conflict-spread)

// A fragment (16x32 f16): lane (m = lane&15, h = lane>>4)
// element e in 0..7  -> K = 8h + e          (contiguous 8 halves)
// element e in 8..15 -> K = 16 + 8h + (e-8) (contiguous 8 halves)
__device__ inline v16h frag_a(const _Float16* lds, int row, int h) {
  const _Float16* p = lds + row * LDH + 8 * h;
  V16HU u;
  u.h8[0] = *(const half8_t*)(p);
  u.h8[1] = *(const half8_t*)(p + 16);
  return u.v;
}

// B fragment (32x16 f16): lane (n = lane&15, h = lane>>4), element e<8 -> K = 8h + e, e>=8 -> K = 16 + 8h + (e-8)
__device__ inline v16h frag_b(const _Float16* lds, int row, int h) {
  const _Float16* p = lds + row * LDH + 8 * h;
  V16HU u;
  u.h8[0] = *(const half8_t*)(p);
  u.h8[1] = *(const half8_t*)(p + 16);
  return u.v;
}

// Stage a 128x32 f32 tile (row-major, leading dim ld) into LDS as f16.
__device__ inline void stage_tile_f32(const float* __restrict__ g, size_t ld,
                                      _Float16* lds, int tid) {
#pragma unroll
  for (int i = 0; i < 4; ++i) {
    int pos = tid + i * 256;        // 1024 float4 positions (128 rows x 8 float4)
    int row = pos >> 3;
    int c4  = pos & 7;
    float4 q = *(const float4*)(g + (size_t)row * ld + c4 * 4);
    half4_t hh = { (_Float16)q.x, (_Float16)q.y, (_Float16)q.z, (_Float16)q.w };
    *(half4_t*)(lds + row * LDH + c4 * 4) = hh;
  }
}

// ---------------------------------------------------------------------------
// GEMM 1: out[b][o][t] = bias[o] + sum_c X[b][t][c] * W[o][c]
//   A = X rows (m = token), B = W rows (n = channel). C[m=t, n=o].
//   Output channel-major [B,256,4096] so the 1024 clustering blocks are
//   contiguous 32768-float slices (all reference reshapes are raw).
// ---------------------------------------------------------------------------
__global__ __launch_bounds__(256) void gemm_xt_w(const float* __restrict__ X,
                                                 const float* __restrict__ W,
                                                 const float* __restrict__ bias,
                                                 float* __restrict__ out) {
  __shared__ __align__(16) _Float16 lA[TILE_M * LDH];
  __shared__ __align__(16) _Float16 lB[TILE_N * LDH];

  const int tid  = threadIdx.x;
  const int lane = tid & 31;
  const int wave = tid >> 5;
  const int h    = lane >> 4;
  const int l16  = lane & 15;
  const int b    = blockIdx.z;
  const int m0   = blockIdx.x * TILE_M;   // token tile
  const int n0   = blockIdx.y * TILE_N;   // channel tile
  const int wm   = (wave & 3) * 32;
  const int wn   = (wave >> 2) * 64;

  const float* Xb = X + (size_t)b * 4096 * 256;
  float*       Ob = out + (size_t)b * 256 * 4096;

  v8f acc[2][4] = {};

  for (int k0 = 0; k0 < 256; k0 += TILE_K) {
    stage_tile_f32(Xb + (size_t)m0 * 256 + k0, 256, lA, tid);
    stage_tile_f32(W  + (size_t)n0 * 256 + k0, 256, lB, tid);
    __syncthreads();

    v16h a0 = frag_a(lA, wm + 0  + l16, h);
    v16h a1 = frag_a(lA, wm + 16 + l16, h);
#pragma unroll
    for (int nf = 0; nf < 4; ++nf) {
      v16h bf = frag_b(lB, wn + nf * 16 + l16, h);
      acc[0][nf] = __builtin_amdgcn_wmma_f32_16x16x32_f16(
          false, a0, false, bf, (short)0, acc[0][nf], false, false);
      acc[1][nf] = __builtin_amdgcn_wmma_f32_16x16x32_f16(
          false, a1, false, bf, (short)0, acc[1][nf], false, false);
    }
    __syncthreads();
  }

#pragma unroll
  for (int mf = 0; mf < 2; ++mf) {
    int tbase = m0 + wm + mf * 16 + 8 * h;   // token
#pragma unroll
    for (int nf = 0; nf < 4; ++nf) {
      int o = n0 + wn + nf * 16 + l16;       // channel
      float bb = bias[o];
      float* dst = Ob + (size_t)o * 4096 + tbase;
      v8f c = acc[mf][nf];
      *(float4*)(dst)     = make_float4(c[0] + bb, c[1] + bb, c[2] + bb, c[3] + bb);
      *(float4*)(dst + 4) = make_float4(c[4] + bb, c[5] + bb, c[6] + bb, c[7] + bb);
    }
  }
}

// ---------------------------------------------------------------------------
// GEMM 2 (conv2): out[b][t][o] = bias[o] + sum_ch W2[o][ch] * Mid[b][ch][t]
//   A = W2 rows (m = o), B = Mid^T staged with transpose (n = t). C[m=o,n=t].
//   Stores token-major d_out layout with contiguous 32B stores.
// ---------------------------------------------------------------------------
__global__ __launch_bounds__(256) void gemm_w_mid(const float* __restrict__ W2,
                                                  const float* __restrict__ bias,
                                                  const float* __restrict__ Mid,
                                                  float* __restrict__ out) {
  __shared__ __align__(16) _Float16 lA[TILE_M * LDH];
  __shared__ __align__(16) _Float16 lB[TILE_N * LDH];

  const int tid  = threadIdx.x;
  const int lane = tid & 31;
  const int wave = tid >> 5;
  const int h    = lane >> 4;
  const int l16  = lane & 15;
  const int b    = blockIdx.z;
  const int m0   = blockIdx.x * TILE_M;   // output-channel tile
  const int n0   = blockIdx.y * TILE_N;   // token tile
  const int wm   = (wave & 3) * 32;
  const int wn   = (wave >> 2) * 64;

  const float* Mb = Mid + (size_t)b * 256 * 4096;

  v8f acc[2][4] = {};

  for (int k0 = 0; k0 < 256; k0 += TILE_K) {
    stage_tile_f32(W2 + (size_t)m0 * 256 + k0, 256, lA, tid);
    // Transposed staging: lB[t_local][kk] = Mid[b][k0+kk][n0+t_local]
#pragma unroll
    for (int i = 0; i < 4; ++i) {
      int pos = tid + i * 256;              // 32 kk-rows x 32 float4 columns
      int kk  = pos >> 5;
      int c4  = pos & 31;
      float4 q = *(const float4*)(Mb + (size_t)(k0 + kk) * 4096 + n0 + c4 * 4);
      lB[(c4 * 4 + 0) * LDH + kk] = (_Float16)q.x;
      lB[(c4 * 4 + 1) * LDH + kk] = (_Float16)q.y;
      lB[(c4 * 4 + 2) * LDH + kk] = (_Float16)q.z;
      lB[(c4 * 4 + 3) * LDH + kk] = (_Float16)q.w;
    }
    __syncthreads();

    v16h a0 = frag_a(lA, wm + 0  + l16, h);
    v16h a1 = frag_a(lA, wm + 16 + l16, h);
#pragma unroll
    for (int nf = 0; nf < 4; ++nf) {
      v16h bf = frag_b(lB, wn + nf * 16 + l16, h);
      acc[0][nf] = __builtin_amdgcn_wmma_f32_16x16x32_f16(
          false, a0, false, bf, (short)0, acc[0][nf], false, false);
      acc[1][nf] = __builtin_amdgcn_wmma_f32_16x16x32_f16(
          false, a1, false, bf, (short)0, acc[1][nf], false, false);
    }
    __syncthreads();
  }

#pragma unroll
  for (int mf = 0; mf < 2; ++mf) {
    int obase = m0 + wm + mf * 16 + 8 * h;   // output channel base (8 consecutive)
    float4 b0 = *(const float4*)(bias + obase);
    float4 b1 = *(const float4*)(bias + obase + 4);
#pragma unroll
    for (int nf = 0; nf < 4; ++nf) {
      int t = n0 + wn + nf * 16 + l16;       // token
      float* dst = out + ((size_t)b * 4096 + t) * 256 + obase;
      v8f c = acc[mf][nf];
      *(float4*)(dst)     = make_float4(c[0] + b0.x, c[1] + b0.y, c[2] + b0.z, c[3] + b0.w);
      *(float4*)(dst + 4) = make_float4(c[4] + b1.x, c[5] + b1.y, c[6] + b1.z, c[7] + b1.w);
    }
  }
}

// ---------------------------------------------------------------------------
// Clustering kernel: one workgroup per b3 block (1024 blocks of 32768 floats).
// All reference reshapes are raw, so each block is a contiguous slice of the
// channel-major conv outputs. 256 threads, ~24KB static LDS.
// ---------------------------------------------------------------------------
__global__ __launch_bounds__(256) void cluster_kernel(
    const float* __restrict__ xh, const float* __restrict__ val,
    const float* __restrict__ feat, const float* __restrict__ convc_w,
    const float* __restrict__ convc_b, const float* __restrict__ alpha_p,
    const float* __restrict__ beta_p, float* __restrict__ outmid) {
  __shared__ float cpool[32 * 16];   // pooled xh   [c3=32][s=16]
  __shared__ float cf[16 * 36];      // centers_feature (bmc remap)
  __shared__ float cenN[16 * 36];    // l2norm(conv_c(centers)) rows (bmc remap)
  __shared__ float cnew[16 * 36];    // centers_new (bmc)
  __shared__ float cen2N[16 * 36];   // l2norm(cen2) rows
  __shared__ float aggb[16 * 36];
  __shared__ float outm[16 * 36];
  __shared__ float cnt[16];
  __shared__ int   mi[1024];
  __shared__ float sv[1024];

  const int tid  = threadIdx.x;
  const int lane = tid & 31;
  const size_t base = (size_t)blockIdx.x * 32768;
  const float* xg = xh + base;
  const float* vg = val + base;
  const float* fg = feat + base;
  float*       og = outmid + base;
  const float alpha = alpha_p[0];
  const float beta  = beta_p[0];
  const float EPS   = 1e-12f;

  // ---- Phase A: adaptive avg pool (8x8 blocks of [32][32]) for xh and feature
  for (int k = tid; k < 1024; k += 256) {
    int which = k >> 9;
    int idx   = k & 511;
    int c  = idx >> 4, s = idx & 15;
    int cw = s >> 2,  cp = s & 3;
    const float* src = (which ? fg : xg) + c * 1024 + (cw * 8) * 32 + cp * 8;
    float sum = 0.f;
#pragma unroll
    for (int w = 0; w < 8; ++w) {
      const float4* r = (const float4*)(src + w * 32);
      float4 q0 = r[0], q1 = r[1];
      sum += q0.x + q0.y + q0.z + q0.w + q1.x + q1.y + q1.z + q1.w;
    }
    sum *= (1.0f / 64.0f);
    if (which == 0) {
      cpool[c * 16 + s] = sum;
    } else {
      int id2 = c * 16 + s;                 // raw reshape [32][16] -> [16][32]
      cf[(id2 >> 5) * 36 + (id2 & 31)] = sum;
    }
  }
  __syncthreads();

  // ---- Phase B: cen = conv_c(centers), remap [32][16] -> [16][32]; zero cnew
  for (int p = tid; p < 512; p += 256) {
    int o = p >> 4, s = p & 15;
    float a = convc_b[o];
    for (int c = 0; c < 32; ++c) a += convc_w[o * 32 + c] * cpool[c * 16 + s];
    int idx = o * 16 + s;
    cenN[(idx >> 5) * 36 + (idx & 31)] = a;
    cnew[(p >> 5) * 36 + (p & 31)] = 0.f;
  }
  __syncthreads();
  if (tid < 16) {  // l2norm rows
    float ss = 0.f;
    for (int c = 0; c < 32; ++c) { float v = cenN[tid * 36 + c]; ss += v * v; }
    float inv = 1.0f / fmaxf(sqrtf(ss), EPS);
    for (int c = 0; c < 32; ++c) cenN[tid * 36 + c] *= inv;
  }
  __syncthreads();

  // ---- Phase C: sim = softmax_m(cenN . l2norm(value)); cnew += sim * feature
  for (int r = 0; r < 4; ++r) {
    int l = r * 256 + tid;
    float vr[32];
    const float4* vp = (const float4*)(vg + (size_t)l * 32);
    float ss = 0.f;
#pragma unroll
    for (int i = 0; i < 8; ++i) {
      float4 q = vp[i];
      vr[i * 4] = q.x; vr[i * 4 + 1] = q.y; vr[i * 4 + 2] = q.z; vr[i * 4 + 3] = q.w;
    }
#pragma unroll
    for (int c = 0; c < 32; ++c) ss += vr[c] * vr[c];
    float inv = 1.0f / fmaxf(sqrtf(ss), EPS);
    float d[16];
#pragma unroll
    for (int m = 0; m < 16; ++m) {
      float a = 0.f;
#pragma unroll
      for (int c = 0; c < 32; ++c) a += cenN[m * 36 + c] * vr[c];
      d[m] = a * inv;
    }
    float mx = d[0];
#pragma unroll
    for (int m = 1; m < 16; ++m) mx = fmaxf(mx, d[m]);
    float se = 0.f;
#pragma unroll
    for (int m = 0; m < 16; ++m) { d[m] = expf(d[m] - mx); se += d[m]; }
    float rs = 1.0f / se;
#pragma unroll
    for (int m = 0; m < 16; ++m) d[m] *= rs;

    float fr[32];
    const float4* fp = (const float4*)(fg + (size_t)l * 32);
#pragma unroll
    for (int i = 0; i < 8; ++i) {
      float4 q = fp[i];
      fr[i * 4] = q.x; fr[i * 4 + 1] = q.y; fr[i * 4 + 2] = q.z; fr[i * 4 + 3] = q.w;
    }
    // Outer-product accumulation with wave32 butterfly reduction:
    // one LDS atomic per wave per (m,c) instead of 32-way serialized atomics.
#pragma unroll
    for (int c = 0; c < 32; ++c) {
      float fc = fr[c];
#pragma unroll
      for (int m = 0; m < 16; ++m) {
        float v = d[m] * fc;
        v += __shfl_xor(v, 16);
        v += __shfl_xor(v, 8);
        v += __shfl_xor(v, 4);
        v += __shfl_xor(v, 2);
        v += __shfl_xor(v, 1);
        if (lane == 0) atomicAdd(&cnew[m * 36 + c], v);
      }
    }
  }
  __syncthreads();

  // ---- Phase D: cen2[m][c] = cnew_flat[c*16+m]; l2norm rows; zero agg/cnt
  for (int p = tid; p < 512; p += 256) {
    int m = p >> 5, c = p & 31;
    int idx = c * 16 + m;
    cen2N[m * 36 + c] = cnew[(idx >> 5) * 36 + (idx & 31)];
    aggb[m * 36 + c]  = 0.f;
  }
  if (tid < 16) cnt[tid] = 0.f;
  __syncthreads();
  if (tid < 16) {
    float ss = 0.f;
    for (int c = 0; c < 32; ++c) { float v = cen2N[tid * 36 + c]; ss += v * v; }
    float inv = 1.0f / fmaxf(sqrtf(ss), EPS);
    for (int c = 0; c < 32; ++c) cen2N[tid * 36 + c] *= inv;
  }
  __syncthreads();

  // ---- Phase E: sim2 = sigmoid(beta + alpha * cen2N . l2norm(xf)); argmax; agg
  for (int r = 0; r < 4; ++r) {
    int l = r * 256 + tid;
    float d[16] = {};
    float ss = 0.f;
    for (int c = 0; c < 32; ++c) {
      float v = xg[c * 1024 + l];   // coalesced: lane index == token index
      ss += v * v;
#pragma unroll
      for (int m = 0; m < 16; ++m) d[m] += cen2N[m * 36 + c] * v;
    }
    float inv = 1.0f / fmaxf(sqrtf(ss), EPS);
    int best = 0;
    float sbest = -1.f;
#pragma unroll
    for (int m = 0; m < 16; ++m) {
      float s = 1.0f / (1.0f + expf(-(beta + alpha * d[m] * inv)));
      if (s > sbest) { sbest = s; best = m; }   // first-max like torch
    }
    mi[l] = best;
    sv[l] = sbest;
    atomicAdd(&cnt[best], 1.0f);
    const float4* fp = (const float4*)(fg + (size_t)l * 32);
#pragma unroll
    for (int i = 0; i < 8; ++i) {
      float4 q = fp[i];
      atomicAdd(&aggb[best * 36 + i * 4 + 0], sbest * q.x);
      atomicAdd(&aggb[best * 36 + i * 4 + 1], sbest * q.y);
      atomicAdd(&aggb[best * 36 + i * 4 + 2], sbest * q.z);
      atomicAdd(&aggb[best * 36 + i * 4 + 3], sbest * q.w);
    }
  }
  __syncthreads();

  // ---- Phase F: outm = (agg + centers_feature) / (count + 1)
  for (int p = tid; p < 512; p += 256) {
    int m = p >> 5, c = p & 31;
    outm[m * 36 + c] = (aggb[m * 36 + c] + cf[m * 36 + c]) / (cnt[m] + 1.0f);
  }
  __syncthreads();

  // ---- Phase G: dispatch back to tokens (raw [L][c] layout == flat slice)
  for (int r = 0; r < 4; ++r) {
    int l = r * 256 + tid;
    int m = mi[l];
    float w = sv[l];
    float4* dp = (float4*)(og + (size_t)l * 32);
#pragma unroll
    for (int i = 0; i < 8; ++i) {
      dp[i] = make_float4(w * outm[m * 36 + i * 4 + 0], w * outm[m * 36 + i * 4 + 1],
                          w * outm[m * 36 + i * 4 + 2], w * outm[m * 36 + i * 4 + 3]);
    }
  }
}

// ---------------------------------------------------------------------------
extern "C" void kernel_launch(void* const* d_in, const int* in_sizes, int n_in,
                              void* d_out, int out_size, void* d_ws, size_t ws_size,
                              hipStream_t stream) {
  const float* x       = (const float*)d_in[0];
  // d_in[1] = H (64), d_in[2] = W (64) — hardcoded.
  const float* conv1_w = (const float*)d_in[3];
  const float* conv1_b = (const float*)d_in[4];
  const float* conv2_w = (const float*)d_in[5];
  const float* conv2_b = (const float*)d_in[6];
  const float* convc_w = (const float*)d_in[7];
  const float* convc_b = (const float*)d_in[8];
  const float* convv_w = (const float*)d_in[9];
  const float* convv_b = (const float*)d_in[10];
  const float* convf_w = (const float*)d_in[11];
  const float* convf_b = (const float*)d_in[12];
  const float* alpha   = (const float*)d_in[13];
  const float* beta    = (const float*)d_in[14];
  float* out = (float*)d_out;

  const size_t NTOT = (size_t)32 * 256 * 4096;   // 33,554,432 floats per tensor
  float* xh   = (float*)d_ws;
  float* val  = xh + NTOT;
  float* feat = val + NTOT;
  float* mid  = feat + NTOT;

  dim3 blk(256);
  dim3 g1(4096 / TILE_M, 256 / TILE_N, 32);      // (32, 2, 32)
  gemm_xt_w<<<g1, blk, 0, stream>>>(x, conv1_w, conv1_b, xh);
  gemm_xt_w<<<g1, blk, 0, stream>>>(x, convv_w, convv_b, val);
  gemm_xt_w<<<g1, blk, 0, stream>>>(x, convf_w, convf_b, feat);

  cluster_kernel<<<dim3(1024), blk, 0, stream>>>(xh, val, feat, convc_w, convc_b,
                                                 alpha, beta, mid);

  dim3 g2(256 / TILE_M, 4096 / TILE_N, 32);      // (2, 32, 32)
  gemm_w_mid<<<g2, blk, 0, stream>>>(conv2_w, conv2_b, mid, out);
}
